// MultiheadSelfAttention_10934986735718
// MI455X (gfx1250) — matrix operand order repaired and checked by reading, not compiled, not run on hardware
//
#include <hip/hip_runtime.h>
#include <hip/hip_bf16.h>

#define EMB 768
#define QKD 64
#define NH  12
#define NB  16
#define SEQ 1024
#define TOK (NB * SEQ)

typedef __attribute__((ext_vector_type(16))) __bf16 v16bf;
typedef __attribute__((ext_vector_type(8)))  float  v8f;

union BFrag {
    v16bf v;
    unsigned int d[8];
    unsigned short u[16];
};

__device__ __forceinline__ unsigned short f32_to_bf16(float x) {
    unsigned int u = __builtin_bit_cast(unsigned int, x);
    u = (u + 0x7FFFu + ((u >> 16) & 1u)) >> 16;   // round-to-nearest-even
    return (unsigned short)u;
}

// CDNA5 async copy: 16 bytes per lane, global -> LDS, tracked on ASYNCcnt.
__device__ __forceinline__ void async_copy_b128(void* lds, const void* g) {
    unsigned lo = (unsigned)(unsigned long long)(uintptr_t)lds;  // LDS byte offset
    unsigned long long ga = (unsigned long long)(uintptr_t)g;
    asm volatile("global_load_async_to_lds_b128 %0, %1, off"
                 :: "v"(lo), "v"(ga) : "memory");
}
#define WAIT_ASYNC_0() asm volatile("s_wait_asynccnt 0x0" ::: "memory")
#define WAIT_ASYNC_3() asm volatile("s_wait_asynccnt 0x3" ::: "memory")
#define WAIT_ASYNC_4() asm volatile("s_wait_asynccnt 0x4" ::: "memory")

// ---------------------------------------------------------------------------
// Kernel 0: one-shot fp32 -> bf16 conversion.
//   zb[t][e]            = z
//   Wt[h][kv][d][e]     = U_qkv[h][kv][e][d]   (transposed for B staging)
//   Umsat[e_out][c]     = U_msa[c][e_out]      (transposed for B staging)
// ---------------------------------------------------------------------------
__global__ __launch_bounds__(256) void convert_kernel(
    const float* __restrict__ z, const float* __restrict__ Uqkv,
    const float* __restrict__ Umsa, unsigned short* __restrict__ zb,
    unsigned short* __restrict__ Wt, unsigned short* __restrict__ Umsat)
{
    const size_t NZ = (size_t)TOK * EMB;
    const size_t NW = (size_t)NH * 3 * EMB * QKD;
    const size_t NU = (size_t)EMB * EMB;
    size_t stride = (size_t)gridDim.x * 256;
    for (size_t i = (size_t)blockIdx.x * 256 + threadIdx.x;
         i < NZ + NW + NU; i += stride) {
        if (i < NZ) {
            zb[i] = f32_to_bf16(z[i]);
        } else if (i < NZ + NW) {
            size_t o = i - NZ;                       // ((g*64)+d)*768 + e
            size_t g = o / (size_t)(EMB * QKD);
            size_t rem = o - g * (EMB * QKD);
            size_t d = rem / EMB, e = rem % EMB;
            Wt[o] = f32_to_bf16(Uqkv[(g * EMB + e) * QKD + d]);
        } else {
            size_t o = i - NZ - NW;                  // eo*768 + c
            size_t eo = o / EMB, c = o % EMB;
            Umsat[o] = f32_to_bf16(Umsa[c * EMB + eo]);
        }
    }
}

// ---------------------------------------------------------------------------
// Kernel 1: QKV projection.  128x64 tile, 8 waves, double-buffered async LDS.
// Q,K stored [kv][h][b][n][d] bf16; V stored transposed [2][h][b][d][n] bf16.
// ---------------------------------------------------------------------------
__global__ __launch_bounds__(256) void qkv_proj_kernel(
    const unsigned short* __restrict__ zb, const unsigned short* __restrict__ Wt,
    unsigned short* __restrict__ qkv_ws)
{
    __shared__ __align__(16) unsigned short Alds[2][128][32];  // [token][k]
    __shared__ __align__(16) unsigned short Blds[2][64][32];   // [d][k]

    const int tid  = threadIdx.x;
    const int lane = tid & 31;
    const int wave = tid >> 5;
    const int lx   = lane & 15;
    const int kh   = lane >> 4;
    const int mblock = blockIdx.x * 128;
    const int nb = blockIdx.y;                 // 0..35
    const int h  = nb / 3, kv = nb % 3;
    const unsigned short* Wb = Wt + (size_t)(h * 3 + kv) * QKD * EMB;  // [d][e]

    // 3 async instructions per thread per tile (2 A-chunks + 1 B-chunk)
    auto stage = [&](int buf, int k0) {
        for (int i = 0; i < 2; ++i) {
            int q = tid + i * 256;             // 512 chunks of 16B (A: 8KB)
            int r = q >> 2, c8 = (q & 3) * 8;
            async_copy_b128(&Alds[buf][r][c8],
                            zb + (size_t)(mblock + r) * EMB + k0 + c8);
        }
        {
            int q = tid;                       // 256 chunks (B: 4KB)
            int d = q >> 2, c8 = (q & 3) * 8;
            async_copy_b128(&Blds[buf][d][c8],
                            Wb + (size_t)d * EMB + k0 + c8);
        }
    };

    v8f acc[4];
    for (int t = 0; t < 4; ++t)
        for (int e = 0; e < 8; ++e) acc[t][e] = 0.f;

    stage(0, 0);
    stage(1, 32);
    for (int k0 = 0; k0 < EMB; k0 += 32) {
        if (k0 + 32 < EMB) WAIT_ASYNC_3(); else WAIT_ASYNC_0();
        __syncthreads();
        const int buf = (k0 >> 5) & 1;

        BFrag a;
        const int m = wave * 16 + lx;
        for (int v = 0; v < 8; ++v) {
            int kk = (v < 4 ? 0 : 16) + kh * 8 + (v & 3) * 2;
            a.d[v] = *(const unsigned int*)&Alds[buf][m][kk];
        }
        for (int t = 0; t < 4; ++t) {
            BFrag bf;
            const int n = t * 16 + lx;
            for (int v = 0; v < 8; ++v) {
                int kk = (v < 4 ? 0 : 16) + kh * 8 + (v & 3) * 2;
                bf.d[v] = *(const unsigned int*)&Blds[buf][n][kk];
            }
            acc[t] = __builtin_amdgcn_wmma_f32_16x16x32_bf16(
                false, a.v, false, bf.v, (short)0, acc[t], false, false);
        }
        __syncthreads();
        if (k0 + 64 < EMB) stage(buf, k0 + 64);
    }

    const size_t base = ((size_t)kv * NH + h) * NB * SEQ * QKD;
    for (int t = 0; t < 4; ++t) {
        for (int r = 0; r < 8; ++r) {
            int mg = mblock + wave * 16 + r + 8 * kh;
            int d  = t * 16 + lx;
            int b  = mg >> 10, n = mg & 1023;
            unsigned short val = f32_to_bf16(acc[t][r]);
            size_t off;
            if (kv < 2) off = base + (((size_t)b * SEQ + n) * QKD + d);
            else        off = base + (((size_t)b * QKD + d) * SEQ + n);  // V^T
            qkv_ws[off] = val;
        }
    }
}

// ---------------------------------------------------------------------------
// Kernel 2: flash attention per (h,b).  4 waves x 16 query rows.
// K/V tiles (32 keys) double-buffered via async LDS, shared by all 4 waves.
// ---------------------------------------------------------------------------
__global__ __launch_bounds__(128) void attn_kernel(
    const unsigned short* __restrict__ qkv_ws, unsigned short* __restrict__ O_ws)
{
    __shared__ __align__(16) unsigned short Klds[2][32][64];  // [key][d]
    __shared__ __align__(16) unsigned short Vlds[2][64][32];  // [d][key]
    __shared__ unsigned short Plds[4][16][32];                // per-wave P

    const int tid  = threadIdx.x;
    const int lane = tid & 31;
    const int wave = tid >> 5;
    const int lx   = lane & 15;
    const int kh   = lane >> 4;
    const int hb = blockIdx.x;                  // 0..191
    const int h  = hb / NB, b = hb % NB;
    const int m_base = blockIdx.y * 64 + wave * 16;

    const size_t blk = (size_t)SEQ * QKD;
    const unsigned short* Qp = qkv_ws + ((size_t)0 * NH + h) * NB * blk + (size_t)b * blk;
    const unsigned short* Kp = qkv_ws + ((size_t)1 * NH + h) * NB * blk + (size_t)b * blk;
    const unsigned short* Vt = qkv_ws + ((size_t)2 * NH + h) * NB * blk + (size_t)b * blk;

    // 4 async instructions per thread per tile (2 K-chunks + 2 V-chunks)
    auto stage = [&](int buf, int j) {
        for (int i = 0; i < 2; ++i) {
            int q = tid + i * 128;             // K tile: 32x64 bf16 = 256 chunks
            int key = q >> 3, c8 = (q & 7) * 8;
            async_copy_b128(&Klds[buf][key][c8],
                            Kp + (size_t)(j * 32 + key) * QKD + c8);
        }
        for (int i = 0; i < 2; ++i) {
            int q = tid + i * 128;             // V tile: 64x32 bf16 = 256 chunks
            int d = q >> 2, c8 = (q & 3) * 8;
            async_copy_b128(&Vlds[buf][d][c8],
                            Vt + (size_t)d * SEQ + j * 32 + c8);
        }
    };

    // Q fragments (16x64 -> two k-steps of 32), loaded once from global
    BFrag aq[2];
    for (int s = 0; s < 2; ++s)
        for (int v = 0; v < 8; ++v) {
            int kk = s * 32 + (v < 4 ? 0 : 16) + kh * 8 + (v & 3) * 2;
            aq[s].d[v] = *(const unsigned int*)&Qp[(size_t)(m_base + lx) * QKD + kk];
        }

    float mrun[8], lrun[8];
    v8f acc[4];
    for (int r = 0; r < 8; ++r) { mrun[r] = -1e30f; lrun[r] = 0.f; }
    for (int t = 0; t < 4; ++t)
        for (int e = 0; e < 8; ++e) acc[t][e] = 0.f;

    stage(0, 0);
    stage(1, 1);
    for (int j = 0; j < SEQ / 32; ++j) {
        if (j + 1 < SEQ / 32) WAIT_ASYNC_4(); else WAIT_ASYNC_0();
        __syncthreads();
        const int buf = j & 1;

        // S = Q K^T for 32 keys (two 16-key C tiles, two k-steps each)
        v8f S[2];
        for (int u = 0; u < 2; ++u)
            for (int e = 0; e < 8; ++e) S[u][e] = 0.f;
        for (int u = 0; u < 2; ++u) {
            const int key = u * 16 + lx;
            for (int s = 0; s < 2; ++s) {
                BFrag bk;
                for (int v = 0; v < 8; ++v) {
                    int kk = s * 32 + (v < 4 ? 0 : 16) + kh * 8 + (v & 3) * 2;
                    bk.d[v] = *(const unsigned int*)&Klds[buf][key][kk];
                }
                S[u] = __builtin_amdgcn_wmma_f32_16x16x32_bf16(
                    false, aq[s].v, false, bk.v, (short)0, S[u], false, false);
            }
        }

        // online softmax (rows live in half-wave groups of 16 lanes)
        for (int r = 0; r < 8; ++r) {
            float s0 = S[0][r] * 0.125f;
            float s1 = S[1][r] * 0.125f;
            float tmax = fmaxf(s0, s1);
            for (int msk = 1; msk < 16; msk <<= 1)
                tmax = fmaxf(tmax, __shfl_xor(tmax, msk, 32));
            float nmax  = fmaxf(mrun[r], tmax);
            float alpha = __expf(mrun[r] - nmax);
            float p0 = __expf(s0 - nmax);
            float p1 = __expf(s1 - nmax);
            float psum = p0 + p1;
            for (int msk = 1; msk < 16; msk <<= 1)
                psum += __shfl_xor(psum, msk, 32);
            lrun[r] = lrun[r] * alpha + psum;
            mrun[r] = nmax;
            for (int t = 0; t < 4; ++t) acc[t][r] *= alpha;
            const int mr = r + 8 * kh;
            Plds[wave][mr][lx]      = f32_to_bf16(p0);
            Plds[wave][mr][16 + lx] = f32_to_bf16(p1);
        }

        // C-layout -> A-layout via per-wave LDS (same-wave DS ops in-order)
        BFrag ap;
        for (int v = 0; v < 8; ++v) {
            int kk = (v < 4 ? 0 : 16) + kh * 8 + (v & 3) * 2;
            ap.d[v] = *(const unsigned int*)&Plds[wave][lx][kk];
        }

        // O += P * V
        for (int t = 0; t < 4; ++t) {
            BFrag bv;
            const int d = t * 16 + lx;
            for (int v = 0; v < 8; ++v) {
                int kk = (v < 4 ? 0 : 16) + kh * 8 + (v & 3) * 2;
                bv.d[v] = *(const unsigned int*)&Vlds[buf][d][kk];
            }
            acc[t] = __builtin_amdgcn_wmma_f32_16x16x32_bf16(
                false, ap.v, false, bv.v, (short)0, acc[t], false, false);
        }
        __syncthreads();
        if (j + 2 < SEQ / 32) stage(buf, j + 2);
    }

    for (int t = 0; t < 4; ++t)
        for (int r = 0; r < 8; ++r) {
            int mg = m_base + r + 8 * kh;
            int d  = t * 16 + lx;
            float o = acc[t][r] / lrun[r];
            O_ws[((size_t)b * SEQ + mg) * (NH * QKD) + h * QKD + d] = f32_to_bf16(o);
        }
}

// ---------------------------------------------------------------------------
// Kernel 3: output projection (fp32 out), same async double-buffered GEMM.
// ---------------------------------------------------------------------------
__global__ __launch_bounds__(256) void out_proj_kernel(
    const unsigned short* __restrict__ O_ws, const unsigned short* __restrict__ Umsat,
    float* __restrict__ out)
{
    __shared__ __align__(16) unsigned short Alds[2][128][32];
    __shared__ __align__(16) unsigned short Blds[2][64][32];

    const int tid  = threadIdx.x;
    const int lane = tid & 31;
    const int wave = tid >> 5;
    const int lx   = lane & 15;
    const int kh   = lane >> 4;
    const int mblock = blockIdx.x * 128;
    const int nblock = blockIdx.y * 64;

    auto stage = [&](int buf, int k0) {
        for (int i = 0; i < 2; ++i) {
            int q = tid + i * 256;
            int r = q >> 2, c8 = (q & 3) * 8;
            async_copy_b128(&Alds[buf][r][c8],
                            O_ws + (size_t)(mblock + r) * EMB + k0 + c8);
        }
        {
            int q = tid;
            int d = q >> 2, c8 = (q & 3) * 8;
            async_copy_b128(&Blds[buf][d][c8],
                            Umsat + (size_t)(nblock + d) * EMB + k0 + c8);
        }
    };

    v8f acc[4];
    for (int t = 0; t < 4; ++t)
        for (int e = 0; e < 8; ++e) acc[t][e] = 0.f;

    stage(0, 0);
    stage(1, 32);
    for (int k0 = 0; k0 < EMB; k0 += 32) {
        if (k0 + 32 < EMB) WAIT_ASYNC_3(); else WAIT_ASYNC_0();
        __syncthreads();
        const int buf = (k0 >> 5) & 1;

        BFrag a;
        const int m = wave * 16 + lx;
        for (int v = 0; v < 8; ++v) {
            int kk = (v < 4 ? 0 : 16) + kh * 8 + (v & 3) * 2;
            a.d[v] = *(const unsigned int*)&Alds[buf][m][kk];
        }
        for (int t = 0; t < 4; ++t) {
            BFrag bf;
            const int n = t * 16 + lx;
            for (int v = 0; v < 8; ++v) {
                int kk = (v < 4 ? 0 : 16) + kh * 8 + (v & 3) * 2;
                bf.d[v] = *(const unsigned int*)&Blds[buf][n][kk];
            }
            acc[t] = __builtin_amdgcn_wmma_f32_16x16x32_bf16(
                false, a.v, false, bf.v, (short)0, acc[t], false, false);
        }
        __syncthreads();
        if (k0 + 64 < EMB) stage(buf, k0 + 64);
    }

    for (int t = 0; t < 4; ++t)
        for (int r = 0; r < 8; ++r) {
            int mg = mblock + wave * 16 + r + 8 * kh;
            int e  = nblock + t * 16 + lx;
            out[(size_t)mg * EMB + e] = acc[t][r];
        }
}

extern "C" void kernel_launch(void* const* d_in, const int* in_sizes, int n_in,
                              void* d_out, int out_size, void* d_ws, size_t ws_size,
                              hipStream_t stream) {
    (void)in_sizes; (void)n_in; (void)out_size; (void)ws_size;
    const float* z    = (const float*)d_in[0];   // [16,1024,768]
    const float* Uqkv = (const float*)d_in[1];   // [12,3,768,64]
    const float* Umsa = (const float*)d_in[2];   // [768,768]
    float* out = (float*)d_out;                  // [16,1024,768]

    unsigned short* zb    = (unsigned short*)d_ws;                       // 12.58M
    unsigned short* Wt    = zb    + (size_t)TOK * EMB;                   // 1.77M
    unsigned short* Umsat = Wt    + (size_t)NH * 3 * EMB * QKD;          // 0.59M
    unsigned short* qkvws = Umsat + (size_t)EMB * EMB;                   // 37.7M
    unsigned short* Ows   = qkvws + (size_t)3 * NH * NB * SEQ * QKD;     // 12.58M

    convert_kernel<<<8192, 256, 0, stream>>>(z, Uqkv, Umsa, zb, Wt, Umsat);

    dim3 g1(TOK / 128, (NH * 3 * QKD) / 64);   // 128 x 36
    qkv_proj_kernel<<<g1, 256, 0, stream>>>(zb, Wt, qkvws);

    dim3 g2(NH * NB, SEQ / 64);                // 192 x 16
    attn_kernel<<<g2, 128, 0, stream>>>(qkvws, Ows);

    dim3 g3(TOK / 128, EMB / 64);              // 128 x 12
    out_proj_kernel<<<g3, 256, 0, stream>>>(Ows, Umsat, out);
}
